// S2VQNetwork_14680198218010
// MI455X (gfx1250) — compile-verified, matrix operand order checked by reading, not run on hardware
//
#include <hip/hip_runtime.h>
#include <hip/hip_bf16.h>

typedef __attribute__((ext_vector_type(16))) _Float16 v16h;
typedef __attribute__((ext_vector_type(8)))  _Float16 v8h;
typedef __attribute__((ext_vector_type(8)))  float    v8f;

#define HID 128
#define NGRAPH 256

// ---------------------------------------------------------------------------
// Weight convert + transpose: W [K, 128] f32 row-major -> Wt [128, Kpad] f16
// (zero-padded along K). Makes B-fragment loads contiguous per lane.
// ---------------------------------------------------------------------------
__global__ void conv_weight(const float* __restrict__ W, _Float16* __restrict__ Wt,
                            int K, int Kpad) {
    int t = blockIdx.x * blockDim.x + threadIdx.x;
    if (t >= 128 * Kpad) return;
    int n = t / Kpad, k = t % Kpad;
    float v = (k < K) ? W[(size_t)k * 128 + n] : 0.0f;
    Wt[(size_t)n * Kpad + k] = (_Float16)v;
}

// ---------------------------------------------------------------------------
// Scatter-add aggregation: agg[dst] += x[src]
// ---------------------------------------------------------------------------
__global__ void scatter7(const int* __restrict__ src, const int* __restrict__ dst,
                         const float* __restrict__ x, float* __restrict__ agg, int E) {
    int t = blockIdx.x * blockDim.x + threadIdx.x;
    if (t >= E * 8) return;
    int e = t >> 3, c = t & 7;
    if (c >= 7) return;
    atomicAdd(&agg[(size_t)dst[e] * 32 + c], x[(size_t)src[e] * 7 + c]);
}

__global__ void scatter128(const int* __restrict__ src, const int* __restrict__ dst,
                           const float* __restrict__ x, float* __restrict__ agg, int E) {
    int t = blockIdx.x * blockDim.x + threadIdx.x;
    if (t >= E * 128) return;
    int e = t >> 7, c = t & 127;
    atomicAdd(&agg[(size_t)dst[e] * 128 + c], x[(size_t)src[e] * 128 + c]);
}

// ---------------------------------------------------------------------------
// GIN pre-MLP combine: h16 = f16((1+eps)*x + agg)
// ---------------------------------------------------------------------------
__global__ void pre_l1(const float* __restrict__ x, const float* __restrict__ agg,
                       const float* __restrict__ eps, _Float16* __restrict__ h16, int N) {
    int t = blockIdx.x * blockDim.x + threadIdx.x;
    if (t >= N * 32) return;
    int i = t >> 5, c = t & 31;
    float v = 0.0f;
    if (c < 7) v = (1.0f + eps[0]) * x[(size_t)i * 7 + c] + agg[t];
    h16[t] = (_Float16)v;
}

__global__ void pre_gen(const float* __restrict__ x, const float* __restrict__ agg,
                        const float* __restrict__ eps, _Float16* __restrict__ h16, int total) {
    int t = blockIdx.x * blockDim.x + threadIdx.x;
    if (t >= total) return;
    h16[t] = (_Float16)((1.0f + eps[0]) * x[t] + agg[t]);
}

// ---------------------------------------------------------------------------
// WMMA GEMM: out[N,128] = A16[N,K] @ Wt16[K,128]^T-layout + bias
// 8 waves/block; wave w computes the 16x16 tile at (blockIdx.x*16, w*16).
// Optional fused ReLU and fused BN statistics (column sum / sumsq atomics).
// ---------------------------------------------------------------------------
template <int K, bool RELU, bool STATS>
__global__ __launch_bounds__(256) void gemm_wmma(
    const _Float16* __restrict__ A, const _Float16* __restrict__ Wt,
    const float* __restrict__ bias, float* __restrict__ out,
    float* __restrict__ stat_sum, float* __restrict__ stat_sumsq) {
    const int lane = threadIdx.x & 31;
    const int wave = threadIdx.x >> 5;       // column tile 0..7
    const int m0   = blockIdx.x * 16;
    const int nlo  = lane & 15;
    const int col  = wave * 16 + nlo;
    const int mRow = m0 + nlo;
    const int koff = (lane >> 4) * 8;        // half-lane K split per ISA layout

    v8f acc = {};
    const _Float16* ap = A  + (size_t)mRow * K + koff;
    const _Float16* bp = Wt + (size_t)col  * K + koff;
#pragma unroll
    for (int kb = 0; kb < K; kb += 32) {
        v8h alo = *(const v8h*)(ap + kb);
        v8h ahi = *(const v8h*)(ap + kb + 16);
        v8h blo = *(const v8h*)(bp + kb);
        v8h bhi = *(const v8h*)(bp + kb + 16);
        v16h av = __builtin_shufflevector(alo, ahi, 0,1,2,3,4,5,6,7,8,9,10,11,12,13,14,15);
        v16h bv = __builtin_shufflevector(blo, bhi, 0,1,2,3,4,5,6,7,8,9,10,11,12,13,14,15);
        acc = __builtin_amdgcn_wmma_f32_16x16x32_f16(
            false, av, false, bv, (short)0, acc, false, false);
    }

    const float bb = bias[col];
    const int mbase = m0 + (lane >> 4) * 8;  // lanes 16-31 hold rows m0+8..m0+15
    float s = 0.0f, ss = 0.0f;
#pragma unroll
    for (int r = 0; r < 8; ++r) {
        float v = acc[r] + bb;
        if (RELU) v = fmaxf(v, 0.0f);
        out[(size_t)(mbase + r) * 128 + col] = v;
        if (STATS) { s += v; ss += v * v; }
    }
    if (STATS) {
        s  += __shfl_xor(s, 16);
        ss += __shfl_xor(ss, 16);
        if (lane < 16) {
            atomicAdd(&stat_sum[col],  s);
            atomicAdd(&stat_sumsq[col], ss);
        }
    }
}

// ---------------------------------------------------------------------------
// BN finalize: fold mean/var/gamma/beta into per-column scale & shift
// ---------------------------------------------------------------------------
__global__ void bn_finalize(const float* __restrict__ sum, const float* __restrict__ sumsq,
                            const float* __restrict__ gamma, const float* __restrict__ beta,
                            float* __restrict__ scale, float* __restrict__ shift, float invN) {
    int c = threadIdx.x;
    float mean = sum[c] * invN;
    float var  = sumsq[c] * invN - mean * mean;
    float rstd = rsqrtf(var + 1e-5f);
    float sc = gamma[c] * rstd;
    scale[c] = sc;
    shift[c] = beta[c] - mean * sc;
}

// BN + ReLU, output f16 (next GEMM operand)
__global__ void bnrelu16(const float* __restrict__ t, const float* __restrict__ scale,
                         const float* __restrict__ shift, _Float16* __restrict__ u16, int total) {
    int i = blockIdx.x * blockDim.x + threadIdx.x;
    if (i >= total) return;
    int c = i & 127;
    u16[i] = (_Float16)fmaxf(t[i] * scale[c] + shift[c], 0.0f);
}

// BN + ReLU, output f32 (layer output / pooling source)
__global__ void bnrelu32(const float* __restrict__ t, const float* __restrict__ scale,
                         const float* __restrict__ shift, float* __restrict__ o, int total) {
    int i = blockIdx.x * blockDim.x + threadIdx.x;
    if (i >= total) return;
    int c = i & 127;
    o[i] = fmaxf(t[i] * scale[c] + shift[c], 0.0f);
}

// ---------------------------------------------------------------------------
// Pooling
// ---------------------------------------------------------------------------
__global__ void pool_accum(const float* __restrict__ x, const int* __restrict__ batch,
                           float* __restrict__ sums, float* __restrict__ cnt, int N) {
    int t = blockIdx.x * blockDim.x + threadIdx.x;
    if (t >= N * 128) return;
    int i = t >> 7, c = t & 127;
    int g = batch[i];
    atomicAdd(&sums[(size_t)g * 128 + c], x[t]);
    if (c == 0) atomicAdd(&cnt[g], 1.0f);
}

__global__ void pool_div(const float* __restrict__ sums, const float* __restrict__ cnt,
                         float* __restrict__ emb) {
    int t = blockIdx.x * blockDim.x + threadIdx.x;
    if (t >= NGRAPH * 128) return;
    int g = t >> 7;
    emb[t] = sums[t] / fmaxf(cnt[g], 1.0f);
}

__global__ void build_combined(const float* __restrict__ x, const float* __restrict__ emb,
                               const int* __restrict__ batch, _Float16* __restrict__ comb, int N) {
    int t = blockIdx.x * blockDim.x + threadIdx.x;
    if (t >= N * 256) return;
    int i = t >> 8, c = t & 255;
    float v = (c < 128) ? x[(size_t)i * 128 + c]
                        : emb[(size_t)batch[i] * 128 + (c - 128)];
    comb[t] = (_Float16)v;
}

// ---------------------------------------------------------------------------
// Final head: out[i] = dot(hq[i,:], Wq2) + bq2 (hq already ReLU'd)
// One wave per node, 4 columns per lane, shuffle reduction.
// ---------------------------------------------------------------------------
__global__ void final_dot(const float* __restrict__ hq, const float* __restrict__ wq2,
                          const float* __restrict__ bq2, float* __restrict__ out, int N) {
    int idx = blockIdx.x * blockDim.x + threadIdx.x;
    int i = idx >> 5, lane = idx & 31;
    if (i >= N) return;
    float p = 0.0f;
#pragma unroll
    for (int c = lane; c < 128; c += 32) p += hq[(size_t)i * 128 + c] * wq2[c];
#pragma unroll
    for (int m = 16; m > 0; m >>= 1) p += __shfl_xor(p, m);
    if (lane == 0) out[i] = p + bq2[0];
}

// ---------------------------------------------------------------------------
// Host launcher
// ---------------------------------------------------------------------------
extern "C" void kernel_launch(void* const* d_in, const int* in_sizes, int n_in,
                              void* d_out, int out_size, void* d_ws, size_t ws_size,
                              hipStream_t stream) {
    (void)n_in; (void)out_size; (void)ws_size;
    const int N = 50000, E = 625000;

    // --- resolve input flattening order (insertion vs pytree-sorted) -------
    int ix = 0, ie = 1, ib = 2, base = 3;
    if (in_sizes[0] == N && in_sizes[1] == 2 * E) {      // sorted top-level: batch, edge_index, params..., x
        ib = 0; ie = 1; base = 2; ix = 2 + 31;
    }
    const float* x_in  = (const float*)d_in[ix];
    const int*   edges = (const int*)d_in[ie];
    const int*   batch = (const int*)d_in[ib];
    const int*   e_src = edges;
    const int*   e_dst = edges + E;

    bool sortedLeaf = (in_sizes[base] != 1);   // first layer leaf: eps(1)=insertion, W1(896)=sorted
    int I_eps, I_W1, I_b1, I_g1, I_be1, I_W2, I_b2, I_g, I_be;
    if (!sortedLeaf) { I_eps=0; I_W1=1; I_b1=2; I_g1=3; I_be1=4; I_W2=5; I_b2=6; I_g=7; I_be=8; }
    else             { I_W1=0; I_W2=1; I_b1=2; I_b2=3; I_be=4; I_be1=5; I_eps=6; I_g=7; I_g1=8; }

    struct Layer { const float *eps,*W1,*b1,*g1,*be1,*W2,*b2,*g,*be; } L[3];
    for (int l = 0; l < 3; ++l) {
        int o = base + l * 9;
        L[l].eps = (const float*)d_in[o + I_eps];
        L[l].W1  = (const float*)d_in[o + I_W1];
        L[l].b1  = (const float*)d_in[o + I_b1];
        L[l].g1  = (const float*)d_in[o + I_g1];
        L[l].be1 = (const float*)d_in[o + I_be1];
        L[l].W2  = (const float*)d_in[o + I_W2];
        L[l].b2  = (const float*)d_in[o + I_b2];
        L[l].g   = (const float*)d_in[o + I_g];
        L[l].be  = (const float*)d_in[o + I_be];
    }
    int qo = base + 27;
    const float* Wq1 = (const float*)d_in[qo + 0];
    const float *Wq2, *bq1, *bq2;
    if (!sortedLeaf) { bq1 = (const float*)d_in[qo+1]; Wq2 = (const float*)d_in[qo+2]; bq2 = (const float*)d_in[qo+3]; }
    else             { Wq2 = (const float*)d_in[qo+1]; bq1 = (const float*)d_in[qo+2]; bq2 = (const float*)d_in[qo+3]; }

    // --- workspace layout ---------------------------------------------------
    char* ws = (char*)d_ws;
    size_t off = 0;
    auto alloc = [&](size_t bytes) -> char* {
        char* p = ws + off;
        off = (off + bytes + 255) & ~(size_t)255;
        return p;
    };
    float*    xbuf   = (float*)   alloc((size_t)N * 128 * 4);  // layer activations
    float*    tbuf   = (float*)   alloc((size_t)N * 128 * 4);  // agg / GEMM out / hq
    _Float16* h16    = (_Float16*)alloc((size_t)N * 128 * 2);  // GEMM f16 operand
    _Float16* comb16 = (_Float16*)alloc((size_t)N * 256 * 2);  // [x | emb[batch]]
    float*    psum   = (float*)   alloc((size_t)NGRAPH * 128 * 4);
    float*    pcnt   = (float*)   alloc((size_t)NGRAPH * 4);
    float*    emb    = (float*)   alloc((size_t)NGRAPH * 128 * 4);
    float*    ssum   = (float*)   alloc(128 * 4);
    float*    ssq    = (float*)   alloc(128 * 4);
    float*    scale  = (float*)   alloc(128 * 4);
    float*    shift  = (float*)   alloc(128 * 4);
    _Float16* w1t[3]; _Float16* w2t[3];
    w1t[0] = (_Float16*)alloc(128 * 32 * 2);
    w1t[1] = (_Float16*)alloc(128 * 128 * 2);
    w1t[2] = (_Float16*)alloc(128 * 128 * 2);
    for (int l = 0; l < 3; ++l) w2t[l] = (_Float16*)alloc(128 * 128 * 2);
    _Float16* wq1t = (_Float16*)alloc(128 * 256 * 2);

    const int TB = 256;
    const float invN = 1.0f / (float)N;

    // --- convert + transpose weights to f16 --------------------------------
    conv_weight<<<(128*32  + TB-1)/TB, TB, 0, stream>>>(L[0].W1, w1t[0], 7,   32);
    conv_weight<<<(128*128 + TB-1)/TB, TB, 0, stream>>>(L[1].W1, w1t[1], 128, 128);
    conv_weight<<<(128*128 + TB-1)/TB, TB, 0, stream>>>(L[2].W1, w1t[2], 128, 128);
    for (int l = 0; l < 3; ++l)
        conv_weight<<<(128*128 + TB-1)/TB, TB, 0, stream>>>(L[l].W2, w2t[l], 128, 128);
    conv_weight<<<(128*256 + TB-1)/TB, TB, 0, stream>>>(Wq1, wq1t, 256, 256);

    const int rowTiles = N / 16;               // 3125, exact
    const int ewTotal  = N * 128;

    for (int l = 0; l < 3; ++l) {
        // scatter-add aggregation into tbuf
        if (l == 0) {
            hipMemsetAsync(tbuf, 0, (size_t)N * 32 * 4, stream);
            scatter7<<<(E*8 + TB-1)/TB, TB, 0, stream>>>(e_src, e_dst, x_in, tbuf, E);
            pre_l1<<<(N*32 + TB-1)/TB, TB, 0, stream>>>(x_in, tbuf, L[0].eps, h16, N);
        } else {
            hipMemsetAsync(tbuf, 0, (size_t)N * 128 * 4, stream);
            scatter128<<<(E*128 + TB-1)/TB, TB, 0, stream>>>(e_src, e_dst, xbuf, tbuf, E);
            pre_gen<<<(ewTotal + TB-1)/TB, TB, 0, stream>>>(xbuf, tbuf, L[l].eps, h16, ewTotal);
        }
        // Linear1 (+ BN stats), BN+ReLU -> f16
        hipMemsetAsync(ssum, 0, 128*4, stream);
        hipMemsetAsync(ssq,  0, 128*4, stream);
        if (l == 0)
            gemm_wmma<32, false, true><<<rowTiles, 256, 0, stream>>>(h16, w1t[0], L[0].b1, tbuf, ssum, ssq);
        else
            gemm_wmma<128, false, true><<<rowTiles, 256, 0, stream>>>(h16, w1t[l], L[l].b1, tbuf, ssum, ssq);
        bn_finalize<<<1, 128, 0, stream>>>(ssum, ssq, L[l].g1, L[l].be1, scale, shift, invN);
        bnrelu16<<<(ewTotal + TB-1)/TB, TB, 0, stream>>>(tbuf, scale, shift, h16, ewTotal);
        // Linear2 (+ BN stats), outer BN+ReLU -> f32 activations
        hipMemsetAsync(ssum, 0, 128*4, stream);
        hipMemsetAsync(ssq,  0, 128*4, stream);
        gemm_wmma<128, false, true><<<rowTiles, 256, 0, stream>>>(h16, w2t[l], L[l].b2, tbuf, ssum, ssq);
        bn_finalize<<<1, 128, 0, stream>>>(ssum, ssq, L[l].g, L[l].be, scale, shift, invN);
        bnrelu32<<<(ewTotal + TB-1)/TB, TB, 0, stream>>>(tbuf, scale, shift, xbuf, ewTotal);
    }

    // --- global mean pool + concat -----------------------------------------
    hipMemsetAsync(psum, 0, (size_t)NGRAPH * 128 * 4, stream);
    hipMemsetAsync(pcnt, 0, (size_t)NGRAPH * 4, stream);
    pool_accum<<<(ewTotal + TB-1)/TB, TB, 0, stream>>>(xbuf, batch, psum, pcnt, N);
    pool_div<<<(NGRAPH*128 + TB-1)/TB, TB, 0, stream>>>(psum, pcnt, emb);
    build_combined<<<(N*256 + TB-1)/TB, TB, 0, stream>>>(xbuf, emb, batch, comb16, N);

    // --- head: [N,256]@[256,128] + ReLU (WMMA), then 128->1 dot ------------
    gemm_wmma<256, true, false><<<rowTiles, 256, 0, stream>>>(comb16, wq1t, bq1, tbuf, ssum, ssq);
    final_dot<<<(N*32 + TB-1)/TB, TB, 0, stream>>>(tbuf, Wq2, bq2, (float*)d_out, N);
}
